// VLADPool2d_60997125538372
// MI455X (gfx1250) — matrix operand order repaired and checked by reading, not compiled, not run on hardware
//
#include <hip/hip_runtime.h>
#include <hip/hip_bf16.h>
#include <math.h>

typedef __attribute__((ext_vector_type(16))) __bf16 v16bf;
typedef __attribute__((ext_vector_type(8)))  __bf16 v8bf;
typedef __attribute__((ext_vector_type(4)))  __bf16 v4bf;
typedef __attribute__((ext_vector_type(8)))  float  v8f;

static constexpr int B_ = 128, C_ = 512, N_ = 512, K_ = 64;

__device__ __forceinline__ v16bf make_frag(const v8bf& a0, const v8bf& a1) {
    v16bf f;
#pragma unroll
    for (int j = 0; j < 8; ++j) { f[j] = a0[j]; f[j + 8] = a1[j]; }
    return f;
}

// ---------------------------------------------------------------------------
// Pass 0a: tiled convert f32 -> bf16, producing BOTH layouts:
//   x_bf[b,c,n] (c-major, GEMM2 A)  and  x_t[b,n,c] (n-major, GEMM1 A)
// 32x32 tile per 256-thread block; LDS transpose so all global R/W coalesced.
// ---------------------------------------------------------------------------
__global__ __launch_bounds__(256)
void vlad_cvt_transpose_kernel(const float* __restrict__ x,
                               __bf16* __restrict__ x_bf,
                               __bf16* __restrict__ x_t) {
    const int tiles_n = N_ / 32, tiles_c = C_ / 32;
    const int blk = blockIdx.x;
    const int b   = blk / (tiles_c * tiles_n);
    const int r   = blk % (tiles_c * tiles_n);
    const int c0  = (r / tiles_n) * 32;
    const int n0  = (r % tiles_n) * 32;

    __shared__ __bf16 tile[32][33];  // +1 pad: dodge LDS bank conflicts

    const int t  = threadIdx.x;
    const int tc = t >> 3;        // c within tile: 0..31
    const int tn = (t & 7) * 4;   // n within tile: 0,4,...,28

    const float4 v = *reinterpret_cast<const float4*>(
        x + ((size_t)b * C_ + c0 + tc) * N_ + n0 + tn);
    const __bf16 e0 = (__bf16)v.x, e1 = (__bf16)v.y,
                 e2 = (__bf16)v.z, e3 = (__bf16)v.w;

    *reinterpret_cast<v4bf*>(x_bf + ((size_t)b * C_ + c0 + tc) * N_ + n0 + tn) =
        (v4bf){e0, e1, e2, e3};

    tile[tc][tn + 0] = e0; tile[tc][tn + 1] = e1;
    tile[tc][tn + 2] = e2; tile[tc][tn + 3] = e3;
    __syncthreads();

    const int on = t >> 3;        // n within tile
    const int oc = (t & 7) * 4;   // c within tile
    const v4bf w = (v4bf){tile[oc + 0][on], tile[oc + 1][on],
                          tile[oc + 2][on], tile[oc + 3][on]};
    *reinterpret_cast<v4bf*>(x_t + ((size_t)b * N_ + n0 + on) * C_ + c0 + oc) = w;
}

// Pass 0b: W (K,C) f32 -> bf16 (tiny: 32K elements).
__global__ __launch_bounds__(256)
void vlad_cvt_w_kernel(const float* __restrict__ W, __bf16* __restrict__ W_bf) {
    const int i = (blockIdx.x * 256 + threadIdx.x) * 4;
    const float4 v = *reinterpret_cast<const float4*>(W + i);
    *reinterpret_cast<v4bf*>(W_bf + i) =
        (v4bf){(__bf16)v.x, (__bf16)v.y, (__bf16)v.z, (__bf16)v.w};
}

// ---------------------------------------------------------------------------
// Kernel 1: logits[b,n,k] = sum_c x[b,c,n]*W[k,c] + bias[k]
// One wave per (batch, 16-row n-tile); 4 accumulators cover K=64.
// 2x-unrolled ping-pong pipeline: loads for the other buffer issue before
// each WMMA bundle; loop-carried values are load destinations (no v_movs).
// ---------------------------------------------------------------------------
template <bool XBF>
__global__ __launch_bounds__(32)
void vlad_logits_kernel(const float* __restrict__ x,
                        const __bf16* __restrict__ x_t,
                        const __bf16* __restrict__ W_bf,
                        const float* __restrict__ bias,
                        float* __restrict__ logits) {
    const int blk  = blockIdx.x;
    const int b    = blk / (N_ / 16);
    const int n0   = (blk % (N_ / 16)) * 16;
    const int lane = threadIdx.x;
    const int lo   = lane & 15, hi = lane >> 4;

    float* __restrict__ Lb = logits + (size_t)b * N_ * K_;

    const __bf16* __restrict__ xrow_t =
        XBF ? (x_t + ((size_t)b * N_ + n0 + lo) * C_) : nullptr;
    const float* __restrict__ xb =
        XBF ? nullptr : (x + (size_t)b * C_ * N_);

    auto loadA = [&](int c0) -> v16bf {
        if constexpr (XBF) {
            // A 16-bit layout: hi=0 -> K {0..7,16..23}; hi=1 -> {8..15,24..31}
            return make_frag(
                *reinterpret_cast<const v8bf*>(xrow_t + c0 + hi * 8),
                *reinterpret_cast<const v8bf*>(xrow_t + c0 + 16 + hi * 8));
        } else {
            v16bf f;
            const int abase = hi * 8;
#pragma unroll
            for (int j = 0; j < 16; ++j) {
                const int cc = c0 + ((j < 8) ? (abase + j) : (16 + abase + (j - 8)));
                f[j] = (__bf16)xb[(size_t)cc * N_ + (n0 + lo)];
            }
            return f;
        }
    };
    auto loadB = [&](int c0, int t) -> v16bf {
        const __bf16* __restrict__ wrow =
            W_bf + (size_t)(t * 16 + lo) * C_ + c0 + hi * 8;
        return make_frag(*reinterpret_cast<const v8bf*>(wrow),
                         *reinterpret_cast<const v8bf*>(wrow + 16));
    };

    v8f acc[4];
#pragma unroll
    for (int t = 0; t < 4; ++t) {
        const float bv = bias[t * 16 + lo];
#pragma unroll
        for (int r = 0; r < 8; ++r) acc[t][r] = bv;
    }

    // Prologue: buffer P holds tile 0.
    v16bf aP = loadA(0);
    v16bf bP[4];
#pragma unroll
    for (int t = 0; t < 4; ++t) bP[t] = loadB(0, t);

    for (int c0 = 0; c0 < C_; c0 += 64) {
        // Load Q = tile c0+32 (always in range: C_ % 64 == 0), then consume P.
        v16bf aQ = loadA(c0 + 32);
        v16bf bQ[4];
#pragma unroll
        for (int t = 0; t < 4; ++t) bQ[t] = loadB(c0 + 32, t);
        if constexpr (XBF) __builtin_prefetch(xrow_t + c0 + 64, 0, 1);
#pragma unroll
        for (int t = 0; t < 4; ++t)
            acc[t] = __builtin_amdgcn_wmma_f32_16x16x32_bf16(
                false, aP, false, bP[t], (short)0, acc[t], false, false);

        // Load P = tile c0+64 (clamped reload on last step), then consume Q.
        const int c2 = (c0 + 64 < C_) ? (c0 + 64) : 0;
        aP = loadA(c2);
#pragma unroll
        for (int t = 0; t < 4; ++t) bP[t] = loadB(c2, t);
#pragma unroll
        for (int t = 0; t < 4; ++t)
            acc[t] = __builtin_amdgcn_wmma_f32_16x16x32_bf16(
                false, aQ, false, bQ[t], (short)0, acc[t], false, false);
    }

#pragma unroll
    for (int t = 0; t < 4; ++t)
#pragma unroll
        for (int r = 0; r < 8; ++r) {
            const int n = n0 + r + 8 * hi;
            Lb[(size_t)n * K_ + t * 16 + lo] = acc[t][r];
        }
}

// ---------------------------------------------------------------------------
// Kernel 2: softmax over the N axis per (b,k); emit a TRANSPOSED (b,k,n) bf16
// + term_c[b,k] = sum_n a*centroids[k,n].
// 256 threads/block: thread = (k, n-group); LDS tree reductions across the
// 4 n-groups.
// ---------------------------------------------------------------------------
__global__ __launch_bounds__(256)
void vlad_softmax_kernel(const float* __restrict__ logits,
                         const float* __restrict__ centroids,
                         __bf16* __restrict__ a_t,
                         float* __restrict__ term_c) {
    const int b = blockIdx.x;
    const int t = threadIdx.x;
    const int k = t & 63;        // column
    const int g = t >> 6;        // n-group: 0..3
    const int nlo = g * (N_ / 4), nhi = nlo + (N_ / 4);
    const float* __restrict__ Lb = logits + (size_t)b * N_ * K_;

    __shared__ float red[4][64];

    float m = -INFINITY;
    for (int n = nlo; n < nhi; ++n) m = fmaxf(m, Lb[n * K_ + k]);
    red[g][k] = m;
    __syncthreads();
    m = fmaxf(fmaxf(red[0][k], red[1][k]), fmaxf(red[2][k], red[3][k]));
    __syncthreads();

    float s = 0.f;
    for (int n = nlo; n < nhi; ++n) s += __expf(Lb[n * K_ + k] - m);
    red[g][k] = s;
    __syncthreads();
    s = red[0][k] + red[1][k] + red[2][k] + red[3][k];
    const float inv = 1.f / s;
    __syncthreads();

    __bf16* __restrict__ Arow = a_t + ((size_t)b * K_ + k) * N_;
    const float* __restrict__ crow = centroids + (size_t)k * C_;
    float tc = 0.f;
    for (int n = nlo; n < nhi; ++n) {
        const float av = __expf(Lb[n * K_ + k] - m) * inv;
        Arow[n] = (__bf16)av;    // per-thread sequential stream
        tc += av * crow[n];
    }
    red[g][k] = tc;
    __syncthreads();
    if (g == 0)
        term_c[b * K_ + k] = red[0][k] + red[1][k] + red[2][k] + red[3][k];
}

// ---------------------------------------------------------------------------
// Kernel 3: vlad[b,c,k] = (sum_n x[b,c,n]*a[b,n,k]) - term_c[b,k]
// Same 2x-unrolled ping-pong; A from x_bf, B from a_t (all b128 loads).
// ---------------------------------------------------------------------------
template <bool XBF>
__global__ __launch_bounds__(32)
void vlad_gemm2_kernel(const float* __restrict__ x,
                       const __bf16* __restrict__ x_bf,
                       const __bf16* __restrict__ a_t,
                       const float* __restrict__ term_c,
                       float* __restrict__ out) {
    const int blk  = blockIdx.x;
    const int b    = blk / (C_ / 16);
    const int c0   = (blk % (C_ / 16)) * 16;
    const int lane = threadIdx.x;
    const int lo   = lane & 15, hi = lane >> 4;

    const __bf16* __restrict__ xrow_bf =
        XBF ? (x_bf + ((size_t)b * C_ + c0 + lo) * N_) : nullptr;
    const float* __restrict__ xrow_f =
        XBF ? nullptr : (x + ((size_t)b * C_ + c0 + lo) * N_);
    const __bf16* __restrict__ Ab = a_t + (size_t)b * K_ * N_;

    auto loadA = [&](int n0) -> v16bf {
        if constexpr (XBF) {
            return make_frag(
                *reinterpret_cast<const v8bf*>(xrow_bf + n0 + hi * 8),
                *reinterpret_cast<const v8bf*>(xrow_bf + n0 + 16 + hi * 8));
        } else {
            v16bf f;
            const int abase = hi * 8;
#pragma unroll
            for (int j = 0; j < 16; ++j) {
                const int nn = n0 + ((j < 8) ? (abase + j) : (16 + abase + (j - 8)));
                f[j] = (__bf16)xrow_f[nn];
            }
            return f;
        }
    };
    auto loadB = [&](int n0, int t) -> v16bf {
        const __bf16* __restrict__ arow =
            Ab + (size_t)(t * 16 + lo) * N_ + n0 + hi * 8;
        return make_frag(*reinterpret_cast<const v8bf*>(arow),
                         *reinterpret_cast<const v8bf*>(arow + 16));
    };

    v8f acc[4] = {};

    v16bf aP = loadA(0);
    v16bf bP[4];
#pragma unroll
    for (int t = 0; t < 4; ++t) bP[t] = loadB(0, t);

    for (int n0 = 0; n0 < N_; n0 += 64) {
        v16bf aQ = loadA(n0 + 32);
        v16bf bQ[4];
#pragma unroll
        for (int t = 0; t < 4; ++t) bQ[t] = loadB(n0 + 32, t);
        if constexpr (XBF) __builtin_prefetch(xrow_bf + n0 + 64, 0, 1);
#pragma unroll
        for (int t = 0; t < 4; ++t)
            acc[t] = __builtin_amdgcn_wmma_f32_16x16x32_bf16(
                false, aP, false, bP[t], (short)0, acc[t], false, false);

        const int n2 = (n0 + 64 < N_) ? (n0 + 64) : 0;
        aP = loadA(n2);
#pragma unroll
        for (int t = 0; t < 4; ++t) bP[t] = loadB(n2, t);
#pragma unroll
        for (int t = 0; t < 4; ++t)
            acc[t] = __builtin_amdgcn_wmma_f32_16x16x32_bf16(
                false, aQ, false, bQ[t], (short)0, acc[t], false, false);
    }

#pragma unroll
    for (int t = 0; t < 4; ++t) {
        const float tc = term_c[b * K_ + t * 16 + lo];
#pragma unroll
        for (int r = 0; r < 8; ++r) {
            const int c = c0 + r + 8 * hi;
            out[((size_t)b * C_ + c) * K_ + t * 16 + lo] = acc[t][r] - tc;
        }
    }
}

// ---------------------------------------------------------------------------
extern "C" void kernel_launch(void* const* d_in, const int* in_sizes, int n_in,
                              void* d_out, int out_size, void* d_ws, size_t ws_size,
                              hipStream_t stream) {
    (void)in_sizes; (void)n_in; (void)out_size;

    const float* x         = (const float*)d_in[0];  // (B, C, N, 1)
    const float* centroids = (const float*)d_in[1];  // (K, C)
    const float* W         = (const float*)d_in[2];  // (K, C)
    const float* bias      = (const float*)d_in[3];  // (K,)
    float*       out       = (float*)d_out;          // (B, C, K, 1)

    auto align256 = [](size_t v) { return (v + 255) & ~(size_t)255; };
    const size_t logits_b = align256((size_t)B_ * N_ * K_ * sizeof(float));   // 16.8 MB
    const size_t at_b     = align256((size_t)B_ * K_ * N_ * sizeof(__bf16));  //  8.4 MB
    const size_t tc_b     = align256((size_t)B_ * K_ * sizeof(float));
    const size_t wbf_b    = align256((size_t)K_ * C_ * sizeof(__bf16));
    const size_t xbf_b    = align256((size_t)B_ * C_ * N_ * sizeof(__bf16));  // 67.1 MB
    const size_t lite     = logits_b + at_b + tc_b + wbf_b;                   // ~25.3 MB
    const size_t full     = lite + 2 * xbf_b;                                 // ~160 MB

    if (ws_size < lite) return;  // cannot run at all

    char* p = (char*)d_ws;
    float*  logits = (float*)p;            p += logits_b;
    __bf16* a_t    = (__bf16*)p;           p += at_b;
    float*  term_c = (float*)p;            p += tc_b;
    __bf16* W_bf   = (__bf16*)p;           p += wbf_b;

    vlad_cvt_w_kernel<<<(K_ * C_) / (256 * 4), 256, 0, stream>>>(W, W_bf);

    if (ws_size >= full) {
        __bf16* x_bf = (__bf16*)p;  p += xbf_b;
        __bf16* x_t  = (__bf16*)p;

        vlad_cvt_transpose_kernel<<<B_ * (C_ / 32) * (N_ / 32), 256, 0, stream>>>(
            x, x_bf, x_t);
        vlad_logits_kernel<true><<<B_ * (N_ / 16), 32, 0, stream>>>(
            x, x_t, W_bf, bias, logits);
        vlad_softmax_kernel<<<B_, 256, 0, stream>>>(logits, centroids, a_t, term_c);
        vlad_gemm2_kernel<true><<<B_ * (C_ / 16), 32, 0, stream>>>(
            x, x_bf, a_t, term_c, out);
    } else {
        vlad_logits_kernel<false><<<B_ * (N_ / 16), 32, 0, stream>>>(
            x, nullptr, W_bf, bias, logits);
        vlad_softmax_kernel<<<B_, 256, 0, stream>>>(logits, centroids, a_t, term_c);
        vlad_gemm2_kernel<false><<<B_ * (C_ / 16), 32, 0, stream>>>(
            x, nullptr, a_t, term_c, out);
    }
}
